// RoPEMultiheadAttention_75471165325819
// MI455X (gfx1250) — matrix operand order repaired and checked by reading, not compiled, not run on hardware
//
#include <hip/hip_runtime.h>
#include <hip/hip_bf16.h>

typedef __bf16 bf16;
typedef __attribute__((ext_vector_type(16))) __bf16 v16bf;
typedef __attribute__((ext_vector_type(8)))  __bf16 v8bf;
typedef __attribute__((ext_vector_type(8)))  float  v8f;
typedef int v4i __attribute__((vector_size(16)));

#define WMMA_BF16(a, b, c) \
  __builtin_amdgcn_wmma_f32_16x16x32_bf16(false, (a), false, (b), (short)0, (c), false, false)

static constexpr int Bsz = 2;
static constexpr int Ssz = 2048;
static constexpr int Esz = 1024;
static constexpr int Hsz = 16;
static constexpr int Dsz = 64;
static constexpr int Mtot = Bsz * Ssz;  // 4096

// ---------------------------------------------------------------------------
// CDNA5 async global->LDS copy (ASYNCcnt) with synchronous fallback.
// Builtin signature (from compiler diagnostic): (v4i AS1*, v4i AS3*, imm, imm)
// ---------------------------------------------------------------------------
#if __has_builtin(__builtin_amdgcn_global_load_async_to_lds_b128) && \
    __has_builtin(__builtin_amdgcn_s_wait_asynccnt)
#define HAVE_ASYNC_LDS 1
#else
#define HAVE_ASYNC_LDS 0
#endif

__device__ __forceinline__ void async_cp16(void* lds_dst, const void* gbl_src) {
#if HAVE_ASYNC_LDS
  __builtin_amdgcn_global_load_async_to_lds_b128(
      (__attribute__((address_space(1))) v4i*)(void*)gbl_src,
      (__attribute__((address_space(3))) v4i*)lds_dst, 0, 0);
#else
  *(int4*)lds_dst = *(const int4*)gbl_src;
#endif
}

#if HAVE_ASYNC_LDS
#define WAIT_ASYNC(n) __builtin_amdgcn_s_wait_asynccnt(n)
#else
#define WAIT_ASYNC(n) ((void)0)
#endif

// ---------------------------------------------------------------------------
// WMMA fragment loaders (CDNA5 16x16x32 bf16 layouts, wave32). Work for both
// global and LDS (generic) pointers; all accesses are 16B-aligned.
//
// A 16x32 (MxK) row-major, ld=ldk: lane m=lane&15; lanes 0-15 hold
//   K={0..7,16..23}, lanes 16-31 hold K={8..15,24..31}.
// B 32x16 (KxN), element (k,n) at base[n*ldk+k]: lane n=lane&15;
//   lanes 0-15 hold K={0..7,16..23}, lanes 16-31 hold K={8..15,24..31}.
// ---------------------------------------------------------------------------
__device__ __forceinline__ v16bf load_a_frag(const bf16* base, int ldk, int lane) {
  int m  = lane & 15;
  int kb = (lane < 16) ? 0 : 8;
  const bf16* p = base + m * ldk + kb;
  v8bf lo = *(const v8bf*)(p);
  v8bf hi = *(const v8bf*)(p + 16);
  v16bf a;
#pragma unroll
  for (int i = 0; i < 8; ++i) { a[i] = lo[i]; a[i + 8] = hi[i]; }
  return a;
}

__device__ __forceinline__ v16bf load_b_frag(const bf16* base, int ldk, int lane) {
  int n  = lane & 15;
  int kb = (lane < 16) ? 0 : 8;
  const bf16* p = base + n * ldk + kb;
  v8bf lo = *(const v8bf*)(p);
  v8bf hi = *(const v8bf*)(p + 16);
  v16bf b;
#pragma unroll
  for (int i = 0; i < 8; ++i) { b[i] = lo[i]; b[i + 8] = hi[i]; }
  return b;
}

// ---------------------------------------------------------------------------
// fp32 -> bf16 conversion
// ---------------------------------------------------------------------------
__global__ __launch_bounds__(256) void cvt_bf16_kernel(const float* __restrict__ in,
                                                       bf16* __restrict__ out, int n) {
  int i = blockIdx.x * 256 + threadIdx.x;
  if (i < n) out[i] = (bf16)in[i];
}

// ---------------------------------------------------------------------------
// LDS-staged, double-buffered GEMM core: block tile 128(M) x 64(N), K-step 64.
// 8 wave32 waves; wave (wm 0..3, wn 0..1) owns a 32x32 sub-tile (4 accums),
// 8 WMMAs per K-stage from LDS fragments. Tiles land in LDS via async copies.
// Row stride 72 elems (144B) keeps 16B alignment + kills bank conflicts.
// ---------------------------------------------------------------------------
static constexpr int TLD = 72;   // padded LDS row stride for 64-elem rows
static constexpr int NKS = Esz / 64;  // 16 K-stages

#define GEMM_STAGE(st, k0)                                                    \
  do {                                                                        \
    _Pragma("unroll")                                                         \
    for (int j = 0; j < 4; ++j) { /* A: 128 rows x 8 chunks */                \
      int idx = tid + 256 * j;                                                \
      int r = idx >> 3, ch = idx & 7;                                         \
      async_cp16(&shA[st][r * TLD + ch * 8],                                  \
                 Arow0 + (size_t)r * Esz + (k0) + ch * 8);                    \
    }                                                                         \
    _Pragma("unroll")                                                         \
    for (int j = 0; j < 2; ++j) { /* W: 64 rows x 8 chunks */                 \
      int idx = tid + 256 * j;                                                \
      int r = idx >> 3, ch = idx & 7;                                         \
      async_cp16(&shW[st][r * TLD + ch * 8],                                  \
                 Wrow0 + (size_t)r * Esz + (k0) + ch * 8);                    \
    }                                                                         \
  } while (0)

#define GEMM_BODY()                                                           \
  const int tid  = threadIdx.x;                                               \
  const int lane = tid & 31;                                                  \
  const int wid  = tid >> 5;                                                  \
  const int wm   = wid & 3;                                                   \
  const int wn   = wid >> 2;                                                  \
  const int rowBase = blockIdx.y * 128;                                       \
  const int colBase = blockIdx.x * 64;                                        \
  const bf16* Arow0 = A + (size_t)rowBase * Esz;                              \
  const bf16* Wrow0 = W + (size_t)colBase * Esz;                              \
  v8f acc00 = {}, acc01 = {}, acc10 = {}, acc11 = {};                         \
  GEMM_STAGE(0, 0);                                                           \
  for (int kk = 0; kk < NKS; ++kk) {                                          \
    const int st = kk & 1;                                                    \
    if (kk + 1 < NKS) { GEMM_STAGE(st ^ 1, (kk + 1) * 64); WAIT_ASYNC(6); }   \
    else              { WAIT_ASYNC(0); }                                      \
    __syncthreads();                                                          \
    const bf16* Abase = &shA[st][(wm * 32) * TLD];                            \
    const bf16* Bbase = &shW[st][(wn * 32) * TLD];                            \
    _Pragma("unroll")                                                         \
    for (int kc = 0; kc < 2; ++kc) {                                          \
      v16bf a0 = load_a_frag(Abase + kc * 32, TLD, lane);                     \
      v16bf a1 = load_a_frag(Abase + 16 * TLD + kc * 32, TLD, lane);          \
      v16bf b0 = load_b_frag(Bbase + kc * 32, TLD, lane);                     \
      v16bf b1 = load_b_frag(Bbase + 16 * TLD + kc * 32, TLD, lane);          \
      acc00 = WMMA_BF16(a0, b0, acc00);                                       \
      acc01 = WMMA_BF16(a0, b1, acc01);                                       \
      acc10 = WMMA_BF16(a1, b0, acc10);                                       \
      acc11 = WMMA_BF16(a1, b1, acc11);                                       \
    }                                                                         \
    __syncthreads();                                                          \
  }                                                                           \
  const int hi8 = (lane >> 4) << 3;                                           \
  const int nl  = lane & 15;

// ---------------------------------------------------------------------------
// QKV projection: z in {0,1,2} selects Wq/Wk/Wv. Q,K -> [B,H,S,D];
// V -> transposed [B,H,D,S] (so attention P@V B-fragments are contiguous).
// ---------------------------------------------------------------------------
__global__ __launch_bounds__(256) void qkv_gemm_kernel(
    const bf16* __restrict__ A,
    const bf16* __restrict__ Wqb, const bf16* __restrict__ Wkb, const bf16* __restrict__ Wvb,
    const float* __restrict__ bq, const float* __restrict__ bk, const float* __restrict__ bv,
    bf16* __restrict__ Qo, bf16* __restrict__ Ko, bf16* __restrict__ Vto) {
  __shared__ __align__(16) bf16 shA[2][128 * TLD];
  __shared__ __align__(16) bf16 shW[2][64 * TLD];
  const int z = blockIdx.z;
  const bf16*  W    = (z == 0) ? Wqb : ((z == 1) ? Wkb : Wvb);
  const float* bias = (z == 0) ? bq  : ((z == 1) ? bk  : bv);

  GEMM_BODY()

#pragma unroll
  for (int ri = 0; ri < 2; ++ri) {
#pragma unroll
    for (int i = 0; i < 8; ++i) {
      const int mg = rowBase + wm * 32 + ri * 16 + i + hi8;
      const int b  = mg >> 11;
      const int s  = mg & (Ssz - 1);
#pragma unroll
      for (int ci = 0; ci < 2; ++ci) {
        const int ng = colBase + wn * 32 + ci * 16 + nl;
        const v8f& acc = ri ? (ci ? acc11 : acc10) : (ci ? acc01 : acc00);
        const float v = acc[i] + bias[ng];
        const int h = ng >> 6;
        const int d = ng & 63;
        if (z == 2) {
          Vto[(((size_t)(b * Hsz + h)) * Dsz + d) * Ssz + s] = (bf16)v;
        } else {
          bf16* dst = (z == 0) ? Qo : Ko;
          dst[(((size_t)(b * Hsz + h)) * Ssz + s) * Dsz + d] = (bf16)v;
        }
      }
    }
  }
}

// ---------------------------------------------------------------------------
// Output projection: d_out[M,E] f32 = Attn_bf16 @ Wo^T + bo
// ---------------------------------------------------------------------------
__global__ __launch_bounds__(256) void oproj_kernel(
    const bf16* __restrict__ A, const bf16* __restrict__ W,
    const float* __restrict__ bias, float* __restrict__ Out) {
  __shared__ __align__(16) bf16 shA[2][128 * TLD];
  __shared__ __align__(16) bf16 shW[2][64 * TLD];

  GEMM_BODY()

#pragma unroll
  for (int ri = 0; ri < 2; ++ri) {
#pragma unroll
    for (int i = 0; i < 8; ++i) {
      const int mg = rowBase + wm * 32 + ri * 16 + i + hi8;
#pragma unroll
      for (int ci = 0; ci < 2; ++ci) {
        const int ng = colBase + wn * 32 + ci * 16 + nl;
        const v8f& acc = ri ? (ci ? acc11 : acc10) : (ci ? acc01 : acc00);
        Out[(size_t)mg * Esz + ng] = acc[i] + bias[ng];
      }
    }
  }
}

// ---------------------------------------------------------------------------
// RoPE on Q and K (bf16 [B,H,S,D], pairs (i, i+32), theta = s*10000^(-2i/64))
// ---------------------------------------------------------------------------
__global__ __launch_bounds__(256) void rope_kernel(bf16* __restrict__ Q, bf16* __restrict__ K) {
  const int idx = blockIdx.x * 256 + threadIdx.x;   // over B*H*S*32 = 2^21
  const int i  = idx & 31;
  const int s  = (idx >> 5) & (Ssz - 1);
  const int bh = idx >> 16;
  const float inv = __expf(-0.28782313662425572f * (float)i);  // ln(1e4)/32
  float c, sn;
  __sincosf((float)s * inv, &c, &sn);
  const size_t base = ((size_t)bh * Ssz + s) * Dsz;

  float q1 = (float)Q[base + i], q2 = (float)Q[base + i + 32];
  Q[base + i]      = (bf16)(q1 * c - q2 * sn);
  Q[base + i + 32] = (bf16)(q2 * c + q1 * sn);

  float k1 = (float)K[base + i], k2 = (float)K[base + i + 32];
  K[base + i]      = (bf16)(k1 * c - k2 * sn);
  K[base + i + 32] = (bf16)(k2 * c + k1 * sn);
}

// ---------------------------------------------------------------------------
// Causal flash attention. Block: (q-tile of 64 rows, one (b,h)).
// K and V tiles staged to LDS with async copies: K drained (<=2) before the
// score phase, V drained (0) only before the P@V phase (overlaps softmax).
// ---------------------------------------------------------------------------
static constexpr int SLD = 65;  // f32 score-tile row stride
static constexpr int PLD = 72;  // bf16 tile row stride (144B, 16B multiple)

__global__ __launch_bounds__(256) void flash_attn_kernel(
    const bf16* __restrict__ Q, const bf16* __restrict__ K,
    const bf16* __restrict__ Vt, bf16* __restrict__ Out) {
  __shared__ __align__(16) float Stile[64 * SLD];
  __shared__ __align__(16) bf16  Ptile[64 * PLD];
  __shared__ __align__(16) bf16  Ktile[64 * PLD];   // [key][d]
  __shared__ __align__(16) bf16  Vtile[64 * PLD];   // [d][key]
  __shared__ float mI[64], lI[64], alphaS[64];

  const int qb = blockIdx.x;
  const int bh = blockIdx.y;
  const bf16* Qh = Q  + (size_t)bh * Ssz * Dsz;
  const bf16* Kh = K  + (size_t)bh * Ssz * Dsz;
  const bf16* Vh = Vt + (size_t)bh * Dsz * Ssz;

  const int tid  = threadIdx.x;
  const int lane = tid & 31;
  const int wid  = tid >> 5;
  const int wm   = wid & 3;
  const int wn   = wid >> 2;
  const int hi8  = (lane >> 4) << 3;
  const int nl   = lane & 15;

  if (tid < 64) { mI[tid] = -3.0e38f; lI[tid] = 0.0f; }

  const bf16* Qrow = Qh + (size_t)(qb * 64 + wm * 16) * Dsz;
  const v16bf qa0 = load_a_frag(Qrow + 0,  Dsz, lane);
  const v16bf qa1 = load_a_frag(Qrow + 32, Dsz, lane);

  v8f o0 = {};
  v8f o1 = {};

  for (int kb = 0; kb <= qb; ++kb) {
    // ---- stage K then V tiles (64 rows x 8 16B-chunks each) ----
#pragma unroll
    for (int j = 0; j < 2; ++j) {
      int idx = tid + 256 * j;
      int r = idx >> 3, ch = idx & 7;
      async_cp16(&Ktile[r * PLD + ch * 8],
                 Kh + (size_t)(kb * 64 + r) * Dsz + ch * 8);
    }
#pragma unroll
    for (int j = 0; j < 2; ++j) {
      int idx = tid + 256 * j;
      int r = idx >> 3, ch = idx & 7;
      async_cp16(&Vtile[r * PLD + ch * 8],
                 Vh + (size_t)r * Ssz + kb * 64 + ch * 8);
    }
    WAIT_ASYNC(2);        // K tile complete (V may still be in flight)
    __syncthreads();

    // ---- phase A: S = (Q K^T)/8, causal mask, spill to LDS ----
    const bf16* K0 = &Ktile[(wn * 32) * PLD];
    const bf16* K1 = &Ktile[(wn * 32 + 16) * PLD];
    v16bf b00 = load_b_frag(K0 + 0,  PLD, lane);
    v16bf b01 = load_b_frag(K0 + 32, PLD, lane);
    v16bf b10 = load_b_frag(K1 + 0,  PLD, lane);
    v16bf b11 = load_b_frag(K1 + 32, PLD, lane);
    v8f s0 = {};
    v8f s1 = {};
    s0 = WMMA_BF16(qa0, b00, s0);
    s0 = WMMA_BF16(qa1, b01, s0);
    s1 = WMMA_BF16(qa0, b10, s1);
    s1 = WMMA_BF16(qa1, b11, s1);

#pragma unroll
    for (int i = 0; i < 8; ++i) {
      const int ml = wm * 16 + i + hi8;
      const int qg = qb * 64 + ml;
      {
        const int cl = wn * 32 + nl;
        float v = s0[i] * 0.125f;
        if (kb * 64 + cl > qg) v = -3.0e38f;
        Stile[ml * SLD + cl] = v;
      }
      {
        const int cl = wn * 32 + 16 + nl;
        float v = s1[i] * 0.125f;
        if (kb * 64 + cl > qg) v = -3.0e38f;
        Stile[ml * SLD + cl] = v;
      }
    }
    __syncthreads();

    // ---- phase B: online softmax per q-row (threads 0..63) ----
    if (tid < 64) {
      const float m_old = mI[tid];
      float mx = m_old;
      for (int c = 0; c < 64; ++c) mx = fmaxf(mx, Stile[tid * SLD + c]);
      const float alpha = __expf(m_old - mx);
      float sum = 0.0f;
      for (int c = 0; c < 64; ++c) {
        const float p = __expf(Stile[tid * SLD + c] - mx);
        sum += p;
        Ptile[tid * PLD + c] = (bf16)p;
      }
      mI[tid] = mx;
      lI[tid] = lI[tid] * alpha + sum;
      alphaS[tid] = alpha;
    }
    WAIT_ASYNC(0);        // V tile complete
    __syncthreads();

    // ---- phase C: O = O*alpha + P @ V ----
#pragma unroll
    for (int i = 0; i < 8; ++i) {
      const float al = alphaS[wm * 16 + i + hi8];
      o0[i] *= al;
      o1[i] *= al;
    }
    const bf16* Prow = Ptile + wm * 16 * PLD;
    v16bf pa0 = load_a_frag(Prow + 0,  PLD, lane);
    v16bf pa1 = load_a_frag(Prow + 32, PLD, lane);
    const bf16* V0 = &Vtile[(wn * 32) * PLD];
    const bf16* V1 = &Vtile[(wn * 32 + 16) * PLD];
    v16bf vb00 = load_b_frag(V0 + 0,  PLD, lane);
    v16bf vb01 = load_b_frag(V0 + 32, PLD, lane);
    v16bf vb10 = load_b_frag(V1 + 0,  PLD, lane);
    v16bf vb11 = load_b_frag(V1 + 32, PLD, lane);
    o0 = WMMA_BF16(pa0, vb00, o0);
    o0 = WMMA_BF16(pa1, vb01, o0);
    o1 = WMMA_BF16(pa0, vb10, o1);
    o1 = WMMA_BF16(pa1, vb11, o1);
    __syncthreads();
  }

  // ---- epilogue: normalize by l_i, write [B,S,E] bf16 ----
  const int b = bh >> 4;
  const int h = bh & 15;
#pragma unroll
  for (int i = 0; i < 8; ++i) {
    const int ml  = wm * 16 + i + hi8;
    const float r = 1.0f / lI[ml];
    const int sg  = qb * 64 + ml;
    const size_t row = ((size_t)(b * Ssz + sg)) * Esz + h * Dsz;
    Out[row + wn * 32 + nl]      = (bf16)(o0[i] * r);
    Out[row + wn * 32 + 16 + nl] = (bf16)(o1[i] * r);
  }
}

// ---------------------------------------------------------------------------
// Host launcher
// ---------------------------------------------------------------------------
extern "C" void kernel_launch(void* const* d_in, const int* in_sizes, int n_in,
                              void* d_out, int out_size, void* d_ws, size_t ws_size,
                              hipStream_t stream) {
  (void)in_sizes; (void)n_in; (void)out_size; (void)ws_size;
  const float* x  = (const float*)d_in[0];
  // d_in[1] = mask (causal; structure known, not read)
  const float* Wq = (const float*)d_in[2];
  const float* bq = (const float*)d_in[3];
  const float* Wk = (const float*)d_in[4];
  const float* bk = (const float*)d_in[5];
  const float* Wv = (const float*)d_in[6];
  const float* bv = (const float*)d_in[7];
  const float* Wo = (const float*)d_in[8];
  const float* bo = (const float*)d_in[9];
  float* out = (float*)d_out;

  char* ws = (char*)d_ws;
  size_t off = 0;
  auto carve = [&](size_t bytes) -> void* {
    void* p = ws + off;
    off = (off + bytes + 255) & ~(size_t)255;
    return p;
  };
  const size_t nX = (size_t)Mtot * Esz;   // 4M elements
  const size_t nW = (size_t)Esz * Esz;    // 1M elements
  bf16* xb  = (bf16*)carve(nX * 2);
  bf16* wqb = (bf16*)carve(nW * 2);
  bf16* wkb = (bf16*)carve(nW * 2);
  bf16* wvb = (bf16*)carve(nW * 2);
  bf16* wob = (bf16*)carve(nW * 2);
  bf16* qb  = (bf16*)carve(nX * 2);       // [B,H,S,D]
  bf16* kb_ = (bf16*)carve(nX * 2);       // [B,H,S,D]
  bf16* vtb = (bf16*)carve(nX * 2);       // [B,H,D,S]
  bf16* ab  = (bf16*)carve(nX * 2);       // attention out [B,S,E]

  cvt_bf16_kernel<<<(int)(nX / 256), 256, 0, stream>>>(x,  xb,  (int)nX);
  cvt_bf16_kernel<<<(int)(nW / 256), 256, 0, stream>>>(Wq, wqb, (int)nW);
  cvt_bf16_kernel<<<(int)(nW / 256), 256, 0, stream>>>(Wk, wkb, (int)nW);
  cvt_bf16_kernel<<<(int)(nW / 256), 256, 0, stream>>>(Wv, wvb, (int)nW);
  cvt_bf16_kernel<<<(int)(nW / 256), 256, 0, stream>>>(Wo, wob, (int)nW);

  qkv_gemm_kernel<<<dim3(Esz / 64, Mtot / 128, 3), 256, 0, stream>>>(
      xb, wqb, wkb, wvb, bq, bk, bv, qb, kb_, vtb);

  rope_kernel<<<(Bsz * Hsz * Ssz * 32) / 256, 256, 0, stream>>>(qb, kb_);

  flash_attn_kernel<<<dim3(Ssz / 64, Bsz * Hsz), 256, 0, stream>>>(qb, kb_, vtb, ab);

  oproj_kernel<<<dim3(Esz / 64, Mtot / 128), 256, 0, stream>>>(ab, wob, bo, out);
}
